// CausalSelfAttention_61658550501969
// MI455X (gfx1250) — matrix operand order repaired and checked by reading, not compiled, not run on hardware
//
#include <hip/hip_runtime.h>

// ---------------------------------------------------------------------------
// Types / helpers
// ---------------------------------------------------------------------------
typedef __attribute__((ext_vector_type(16))) __bf16 v16bf;
typedef __attribute__((ext_vector_type(8)))  float  v8f;
typedef __attribute__((ext_vector_type(4)))  unsigned int u32x4;
typedef __attribute__((ext_vector_type(8)))  unsigned int u32x8;

union Frag16 { v16bf v; unsigned int u[8]; };

__device__ __forceinline__ unsigned short f2bf(float f) {
  unsigned int u = __float_as_uint(f);
  u += 0x7FFFu + ((u >> 16) & 1u);          // round-to-nearest-even
  return (unsigned short)(u >> 16);
}
__device__ __forceinline__ float bf2f(unsigned short h) {
  return __uint_as_float(((unsigned int)h) << 16);
}

// Async global->LDS copy (16B per lane), ASYNCcnt-tracked, no VGPR bounce.
__device__ __forceinline__ void async_g2l_b128(const void* g, const void* lds) {
  unsigned l = (unsigned)(unsigned long long)lds;   // aperture low 32 bits = LDS offset
  asm volatile("global_load_async_to_lds_b128 %0, %1, off"
               :: "v"(l), "v"(g) : "memory");
}

// Tensor Data Mover: 2D tile load, 32 rows x 64 bf16, row stride 64 elems in
// memory; LDS padded by 1 dword per 128B row -> effective LDS stride 66 bf16.
__device__ __forceinline__ void tdm_load_k_tile(const unsigned short* gsrc,
                                                const unsigned short* ldsdst) {
  unsigned long long ga = (unsigned long long)gsrc;
  u32x4 g0;
  g0[0] = 1u;                                             // count=1 (user mode)
  g0[1] = (unsigned)(unsigned long long)ldsdst;           // lds_addr
  g0[2] = (unsigned)ga;                                   // global_addr[31:0]
  g0[3] = ((unsigned)(ga >> 32) & 0x01FFFFFFu) | 0x80000000u; // addr[56:32] | type=2
  u32x8 g1;
  g1[0] = (1u << 16)        // data_size = 2B
        | (1u << 20)        // pad_enable
        | (4u << 22)        // pad_interval: 32 dwords (= 128B = one 64-elem row)
        | (0u << 25);       // pad_amount: 1 dword (2 bf16) -> LDS stride 66
  g1[1] = 64u << 16;        // tensor_dim0[15:0] = 64 (atomic_barrier_addr = 0)
  g1[2] = 2048u << 16;      // tensor_dim0[31:16]=0 | tensor_dim1[15:0] = 2048
  g1[3] = 64u << 16;        // tensor_dim1[31:16]=0 | tile_dim0 = 64
  g1[4] = 32u;              // tile_dim1 = 32 | tile_dim2 = 0
  g1[5] = 64u;              // tensor_dim0_stride[31:0] = 64 elems
  g1[6] = 0u;               // dim0_stride hi | dim1_stride lo
  g1[7] = 0u;               // dim1_stride hi
  u32x4 gz; gz[0] = 0u; gz[1] = 0u; gz[2] = 0u; gz[3] = 0u;
  asm volatile("tensor_load_to_lds %0, %1, %2, %3"
               :: "s"(g0), "s"(g1), "s"(gz), "s"(gz) : "memory");
}

// ---------------------------------------------------------------------------
// 1) fp32 -> bf16 conversion
// ---------------------------------------------------------------------------
__global__ void cvt_f32_bf16(const float* __restrict__ src,
                             unsigned short* __restrict__ dst, int n) {
  int i = blockIdx.x * blockDim.x + threadIdx.x;
  if (i < n) dst[i] = f2bf(src[i]);
}

// ---------------------------------------------------------------------------
// 2) Fused QKV projection: Y = X @ W^T  (M=4096, K=1024, N=3072)
//    Async double-buffered LDS staging, 64x64 tile, 8 waves, K-steps of 32.
//    LDS row stride 40 bf16 (80B): 16B-aligned for async B128 and
//    bank-conflict-free for the fragment gathers.
// ---------------------------------------------------------------------------
__global__ __launch_bounds__(256)
void gemm_qkv(const unsigned short* __restrict__ xbf,
              const unsigned short* __restrict__ wqbf,
              const unsigned short* __restrict__ wkvbf,
              unsigned short* __restrict__ qbuf,
              unsigned short* __restrict__ kbuf,
              unsigned short* __restrict__ vbuf) {
  __shared__ unsigned short sA[2 * 64 * 40];
  __shared__ unsigned short sB[2 * 64 * 40];

  const int t    = threadIdx.x;
  const int lane = t & 31;
  const int w    = t >> 5;        // wave 0..7
  const int half = lane >> 4;
  const int l16  = lane & 15;

  const int m0  = blockIdx.y * 64;
  const int n0  = blockIdx.x * 64;
  const int mt  = w & 3;          // wave's 16-row subtile
  const int ntb = (w >> 2) * 2;   // wave's two 16-col subtiles

  v8f acc0 = {}; v8f acc1 = {};

  const int ldr = t >> 2;         // staging row 0..63
  const int ldq = t & 3;          // staging 8-elem quarter

  const unsigned short* gA = xbf + (m0 + ldr) * 1024 + ldq * 8;
  const int nrow = n0 + ldr;
  const unsigned short* gB = ((nrow < 1024) ? (wqbf + nrow * 1024)
                                            : (wkvbf + (nrow - 1024) * 1024)) + ldq * 8;
  unsigned short* dA = sA + ldr * 40 + ldq * 8;
  unsigned short* dB = sB + ldr * 40 + ldq * 8;

  // Prologue: stage tile 0 into buffer 0 asynchronously.
  async_g2l_b128(gA, dA);
  async_g2l_b128(gB, dB);

  for (int kt = 0; kt < 32; ++kt) {
    const int buf = kt & 1;
    if (kt + 1 < 32) {
      // issue next tile into the other buffer, allow it to stay in flight
      async_g2l_b128(gA + (kt + 1) * 32, dA + (buf ^ 1) * 2560);
      async_g2l_b128(gB + (kt + 1) * 32, dB + (buf ^ 1) * 2560);
      asm volatile("s_wait_asynccnt 0x2" ::: "memory");
    } else {
      asm volatile("s_wait_asynccnt 0x0" ::: "memory");
    }
    __syncthreads();

    const unsigned short* cA = sA + buf * 2560;
    const unsigned short* cB = sB + buf * 2560;

    // A fragment: 16x32 bf16 ISA layout (lane m=l16, k pairs per VGPR/half)
    Frag16 a;
#pragma unroll
    for (int v = 0; v < 8; ++v) {
      int koff = ((v < 4) ? 0 : 16) + half * 8 + (v & 3) * 2;
      a.u[v] = *(const unsigned int*)(cA + (mt * 16 + l16) * 40 + koff);
    }
    // B fragments: 32x16 bf16 (lane n=l16, same k order as A)
    Frag16 b0, b1;
#pragma unroll
    for (int v = 0; v < 8; ++v) {
      int koff = ((v < 4) ? 0 : 16) + half * 8 + (v & 3) * 2;
      b0.u[v] = *(const unsigned int*)(cB + ( ntb      * 16 + l16) * 40 + koff);
      b1.u[v] = *(const unsigned int*)(cB + ((ntb + 1) * 16 + l16) * 40 + koff);
    }
    acc0 = __builtin_amdgcn_wmma_f32_16x16x32_bf16(false, a.v, false, b0.v, (short)0, acc0, false, false);
    acc1 = __builtin_amdgcn_wmma_f32_16x16x32_bf16(false, a.v, false, b1.v, (short)0, acc1, false, false);
    __syncthreads();
  }

  // Epilogue: scatter to q/k/v (bf16)
#pragma unroll
  for (int j = 0; j < 2; ++j) {
    v8f accv = j ? acc1 : acc0;
#pragma unroll
    for (int v = 0; v < 8; ++v) {
      int m    = v + 8 * half;                     // C-layout row
      int grow = m0 + mt * 16 + m;                 // global M row = b*2048 + s
      int bi   = grow >> 11;
      int s    = grow & 2047;
      int n    = n0 + (ntb + j) * 16 + l16;
      unsigned short val = f2bf(accv[v]);
      if (n < 1024) {
        int h = n >> 6, d = n & 63;
        qbuf[(((bi * 16 + h) * 2048) + s) * 64 + d] = val;
      } else {
        int n2 = n - 1024;
        int h = n2 >> 7, rem = n2 & 127;
        int d = rem >> 1;
        unsigned short* dst = (rem & 1) ? vbuf : kbuf;
        dst[(((bi * 16 + h) * 2048) + s) * 64 + d] = val;
      }
    }
  }
}

// ---------------------------------------------------------------------------
// 3) RoPE in-place on q,k.  One thread per (row, pair p<32).
// ---------------------------------------------------------------------------
__global__ void rope_qk(unsigned short* __restrict__ qbuf,
                        unsigned short* __restrict__ kbuf) {
  int idx = blockIdx.x * blockDim.x + threadIdx.x;   // < 65536*32
  int row = idx >> 5;                                // (b*H+h)*S + s
  int p   = idx & 31;
  int s   = row & 2047;
  float inv = __powf(10000.0f, -(float)p * (1.0f / 32.0f));
  float ang = (float)s * inv;
  float c  = __cosf(ang);
  float sn = __sinf(ang);
  {
    float x1 = bf2f(qbuf[row * 64 + p]);
    float x2 = bf2f(qbuf[row * 64 + p + 32]);
    qbuf[row * 64 + p]      = f2bf(x1 * c - x2 * sn);
    qbuf[row * 64 + p + 32] = f2bf(x2 * c + x1 * sn);
  }
  {
    float x1 = bf2f(kbuf[row * 64 + p]);
    float x2 = bf2f(kbuf[row * 64 + p + 32]);
    kbuf[row * 64 + p]      = f2bf(x1 * c - x2 * sn);
    kbuf[row * 64 + p + 32] = f2bf(x2 * c + x1 * sn);
  }
}

// ---------------------------------------------------------------------------
// 4) Flash attention.  WG = 4 waves, 64 query rows (16/wave).
//    K tile staged by the Tensor Data Mover (TDM) with hardware padding;
//    V staged manually (transposed).  Online softmax, fp32 WMMA accum.
// ---------------------------------------------------------------------------
__global__ __launch_bounds__(128)
void attn_fwd(const unsigned short* __restrict__ qbuf,
              const unsigned short* __restrict__ kbuf,
              const unsigned short* __restrict__ vbuf,
              float* __restrict__ out) {
  const int S = 2048;
  __shared__ unsigned short sK [32 * 66];      // keys x hd; stride 66 = TDM pad
  __shared__ unsigned short sVT[64 * 34];      // hd x keys (transposed)
  __shared__ unsigned short sP [4 * 16 * 34];  // per-wave P scratch (C->A relayout)

  const int t    = threadIdx.x;
  const int lane = t & 31;
  const int w    = t >> 5;        // wave 0..3
  const int half = lane >> 4;
  const int l16  = lane & 15;

  const int bh   = blockIdx.x >> 5;    // (b*16+h)
  const int qblk = blockIdx.x & 31;
  const int qb   = qblk * 64;
  const int b    = bh >> 4;
  const int h    = bh & 15;

  const unsigned short* Q = qbuf + (size_t)bh * S * 64;
  const unsigned short* K = kbuf + (size_t)bh * S * 64;
  const unsigned short* V = vbuf + (size_t)bh * S * 64;

  // Preload this wave's Q fragments (16 rows x hd[0:32] and hd[32:64])
  Frag16 aq0, aq1;
  {
    const unsigned short* qrow = Q + (qb + w * 16 + l16) * 64;
#pragma unroll
    for (int v = 0; v < 8; ++v) {
      int koff = ((v < 4) ? 0 : 16) + half * 8 + (v & 3) * 2;
      aq0.u[v] = *(const unsigned int*)(qrow + koff);
      aq1.u[v] = *(const unsigned int*)(qrow + 32 + koff);
    }
  }

  v8f acc[4] = { v8f{}, v8f{}, v8f{}, v8f{} };
  float rowmax[8], rowsum[8];
#pragma unroll
  for (int v = 0; v < 8; ++v) { rowmax[v] = -__builtin_inff(); rowsum[v] = 0.0f; }

  const float scale = 0.125f;          // 1/sqrt(64)
  const int   nj    = qb / 32 + 2;     // causal: key blocks covering row qb+63
  const int   ldr   = t >> 2;          // staging: key 0..31
  const int   ldq   = t & 3;           // staging: hd quarter (16 elems)

  for (int j = 0; j < nj; ++j) {
    const int kb = j * 32;

    // K tile: one TDM descriptor issued by wave 0 (EXEC-independent DMA)
    if (w == 0) tdm_load_k_tile(K + (size_t)kb * 64, sK);

    { // V tile transposed: sVT[hd][key]  (overlaps with TDM transfer)
      const unsigned short* g = V + (kb + ldr) * 64 + ldq * 16;
#pragma unroll
      for (int i = 0; i < 16; ++i) sVT[(ldq * 16 + i) * 34 + ldr] = g[i];
    }
    if (j + 1 < nj)   // prefetch next V tile toward L2
      __builtin_prefetch(V + (size_t)(kb + 32) * 64 + ldr * 64, 0, 1);

    if (w == 0) __builtin_amdgcn_s_wait_tensorcnt(0);
    __syncthreads();

    // ---- S = Q K^T  (contraction over hd=64, two K=32 WMMA steps) ----
    v8f s0 = {}, s1 = {};
    {
      Frag16 b0, b1;
#pragma unroll
      for (int v = 0; v < 8; ++v) {
        int koff = ((v < 4) ? 0 : 16) + half * 8 + (v & 3) * 2;
        b0.u[v] = *(const unsigned int*)(sK + ( l16     ) * 66 + koff);
        b1.u[v] = *(const unsigned int*)(sK + (16 + l16 ) * 66 + koff);
      }
      s0 = __builtin_amdgcn_wmma_f32_16x16x32_bf16(false, aq0.v, false, b0.v, (short)0, s0, false, false);
      s1 = __builtin_amdgcn_wmma_f32_16x16x32_bf16(false, aq0.v, false, b1.v, (short)0, s1, false, false);
#pragma unroll
      for (int v = 0; v < 8; ++v) {
        int koff = 32 + ((v < 4) ? 0 : 16) + half * 8 + (v & 3) * 2;
        b0.u[v] = *(const unsigned int*)(sK + ( l16     ) * 66 + koff);
        b1.u[v] = *(const unsigned int*)(sK + (16 + l16 ) * 66 + koff);
      }
      s0 = __builtin_amdgcn_wmma_f32_16x16x32_bf16(false, aq1.v, false, b0.v, (short)0, s0, false, false);
      s1 = __builtin_amdgcn_wmma_f32_16x16x32_bf16(false, aq1.v, false, b1.v, (short)0, s1, false, false);
    }

    // ---- causal mask + online softmax (per row m = v + 8*half) ----
    float p0[8], p1[8], alpha[8];
#pragma unroll
    for (int v = 0; v < 8; ++v) {
      int qi  = qb + w * 16 + v + 8 * half;
      float sv0 = (kb + l16      <= qi) ? s0[v] * scale : -__builtin_inff();
      float sv1 = (kb + 16 + l16 <= qi) ? s1[v] * scale : -__builtin_inff();
      float tmax = fmaxf(sv0, sv1);
#pragma unroll
      for (int off = 1; off < 16; off <<= 1)
        tmax = fmaxf(tmax, __shfl_xor(tmax, off, 32));
      float nm = fmaxf(rowmax[v], tmax);
      float al = __expf(rowmax[v] - nm);   // first tile: exp(-inf)=0
      rowmax[v] = nm;
      alpha[v]  = al;
      float e0 = __expf(sv0 - nm);
      float e1 = __expf(sv1 - nm);
      float ts = e0 + e1;
#pragma unroll
      for (int off = 1; off < 16; off <<= 1)
        ts += __shfl_xor(ts, off, 32);
      rowsum[v] = rowsum[v] * al + ts;
      p0[v] = e0; p1[v] = e1;
    }

    // rescale running accumulator
#pragma unroll
    for (int nt = 0; nt < 4; ++nt)
#pragma unroll
      for (int v = 0; v < 8; ++v)
        acc[nt][v] *= alpha[v];

    // ---- P: C-layout -> A-layout via per-wave LDS scratch ----
    unsigned short* pw = sP + w * 16 * 34;
#pragma unroll
    for (int v = 0; v < 8; ++v) {
      int m = v + 8 * half;
      pw[m * 34 + l16]      = f2bf(p0[v]);
      pw[m * 34 + 16 + l16] = f2bf(p1[v]);
    }
    Frag16 ap;
#pragma unroll
    for (int v = 0; v < 8; ++v) {
      int koff = ((v < 4) ? 0 : 16) + half * 8 + (v & 3) * 2;
      ap.u[v] = *(const unsigned int*)(pw + l16 * 34 + koff);
    }

    // ---- O += P @ V  (contraction over 32 keys, 4 hd subtiles) ----
#pragma unroll
    for (int nt = 0; nt < 4; ++nt) {
      Frag16 bv;
#pragma unroll
      for (int v = 0; v < 8; ++v) {
        int koff = ((v < 4) ? 0 : 16) + half * 8 + (v & 3) * 2;
        bv.u[v] = *(const unsigned int*)(sVT + (nt * 16 + l16) * 34 + koff);
      }
      acc[nt] = __builtin_amdgcn_wmma_f32_16x16x32_bf16(false, ap.v, false, bv.v, (short)0, acc[nt], false, false);
    }
    __syncthreads();
  }

  // ---- normalize + write (B,S,D) fp32 ----
#pragma unroll
  for (int nt = 0; nt < 4; ++nt) {
#pragma unroll
    for (int v = 0; v < 8; ++v) {
      int m    = v + 8 * half;
      int srow = qb + w * 16 + m;
      int col  = h * 64 + nt * 16 + l16;
      out[((size_t)b * S + srow) * 1024 + col] = acc[nt][v] / rowsum[v];
    }
  }
}

// ---------------------------------------------------------------------------
// Host launch
// ---------------------------------------------------------------------------
extern "C" void kernel_launch(void* const* d_in, const int* in_sizes, int n_in,
                              void* d_out, int out_size, void* d_ws, size_t ws_size,
                              hipStream_t stream) {
  const float* x   = (const float*)d_in[0];   // (2,2048,1024)
  const float* Wq  = (const float*)d_in[1];   // (1024,1024)
  const float* Wkv = (const float*)d_in[2];   // (2048,1024)
  // d_in[3] = causal mask (known tril) -> not needed
  float* out = (float*)d_out;

  const size_t NX   = 2u * 2048u * 1024u;     // 4194304
  const size_t NWQ  = 1024u * 1024u;          // 1048576
  const size_t NWKV = 2048u * 1024u;          // 2097152
  const size_t NQKV = 2u * 16u * 2048u * 64u; // 4194304 per buffer

  size_t off = 0;
  unsigned short* xbf   = (unsigned short*)((char*)d_ws + off); off += NX   * 2;
  unsigned short* wqbf  = (unsigned short*)((char*)d_ws + off); off += NWQ  * 2;
  unsigned short* wkvbf = (unsigned short*)((char*)d_ws + off); off += NWKV * 2;
  unsigned short* qbuf  = (unsigned short*)((char*)d_ws + off); off += NQKV * 2;
  unsigned short* kbuf  = (unsigned short*)((char*)d_ws + off); off += NQKV * 2;
  unsigned short* vbuf  = (unsigned short*)((char*)d_ws + off); off += NQKV * 2;
  if (ws_size < off) return;   // ~38 MB required

  cvt_f32_bf16<<<(NX   + 255) / 256, 256, 0, stream>>>(x,   xbf,   (int)NX);
  cvt_f32_bf16<<<(NWQ  + 255) / 256, 256, 0, stream>>>(Wq,  wqbf,  (int)NWQ);
  cvt_f32_bf16<<<(NWKV + 255) / 256, 256, 0, stream>>>(Wkv, wkvbf, (int)NWKV);

  gemm_qkv<<<dim3(48, 64), 256, 0, stream>>>(xbf, wqbf, wkvbf, qbuf, kbuf, vbuf);

  rope_qk<<<(2u * 16u * 2048u * 32u) / 256u, 256, 0, stream>>>(qbuf, kbuf);

  attn_fwd<<<2 * 16 * (2048 / 64), 128, 0, stream>>>(qbuf, kbuf, vbuf, out);
}
